// TreeLSTM_31121333027309
// MI455X (gfx1250) — compile-verified, matrix operand order checked by reading, not run on hardware
//
#include <hip/hip_runtime.h>
#include <hip/hip_bf16.h>
#include <cstddef>

// ---------------------------------------------------------------------------
// Child-sum TreeLSTM, fused per-level kernel for gfx1250 (MI455X).
//   - bf16 WMMA (v_wmma_f32_16x16x32_bf16) for all GEMMs, f32 accumulate
//   - weights pre-packed once per launch into bf16 WMMA B-fragment layout
//     (256 KB, L2-resident) -> hot loop has zero f32->bf16 weight converts
//   - branch-free gate activations via native v_exp_f32 / v_rcp_f32
//   - heap-indexed h/c state in d_ws:  h[B][255][128], c[B][255][128]  (f32)
//     ws_size >= 2*4096*255*128*4 + 262144 bytes (~1.07 GB)
//   - 1 prepack + 8 dependent level launches (depth 7 .. 0) on `stream`
// ---------------------------------------------------------------------------

typedef __attribute__((ext_vector_type(16))) __bf16 v16bf;
typedef __attribute__((ext_vector_type(8)))  __bf16 v8bf;
typedef __attribute__((ext_vector_type(8)))  float  v8f;

#define NB      4096      // batch (trees)
#define NNODES  255       // nodes per tree (heap layout)
#define HID     128

// Packed-weight tile table (64 tiles of 16 output columns, K=128 each):
//   tiles  0..23 : W_iou   (384 cols)
//   tiles 24..47 : U_iou   (384 cols)
//   tiles 48..55 : W_f     (128 cols)
//   tiles 56..63 : U_f     (128 cols)
// Layout: packed[((tile*4 + kstep)*32 + lane) * 16 + j]  (bf16)
#define NPACK_TILES 64

#define LOG2E  1.4426950408889634f

// Branch-free activations on the native exp2 / rcp transcendental path.
static __device__ __forceinline__ float sigm(float x) {
  return __builtin_amdgcn_rcpf(1.0f + __builtin_amdgcn_exp2f(-LOG2E * x));
}
static __device__ __forceinline__ float tanh_fast(float x) {
  // tanh(x) = 1 - 2/(exp(2x)+1); rcp(inf)=0 and exp2(-big)=0 give exact +-1 limits
  return 1.0f - 2.0f * __builtin_amdgcn_rcpf(1.0f + __builtin_amdgcn_exp2f(2.0f * LOG2E * x));
}

// A-fragment (16x32 bf16, MxK) from LDS row-major [16][128] bf16.
// wave32 layout: lane -> M = lane&15, half = lane>>4;
//   elems 0..7  = S[M][kbase + half*8 + j]
//   elems 8..15 = S[M][kbase + 16 + half*8 + j]
static __device__ __forceinline__ v16bf load_a_frag(const __bf16* S, int kbase) {
  const int lane = threadIdx.x & 31;
  const __bf16* p = S + (lane & 15) * HID + kbase + (lane >> 4) * 8;
  v8bf lo = *(const v8bf*)p;
  v8bf hi = *(const v8bf*)(p + 16);
  return __builtin_shufflevector(lo, hi, 0,1,2,3,4,5,6,7,8,9,10,11,12,13,14,15);
}

// B-fragment from pre-packed bf16 weights: one contiguous 32B load per lane.
static __device__ __forceinline__ v16bf load_b_frag(const __bf16* P, int tile, int s) {
  const int lane = threadIdx.x & 31;
  return *(const v16bf*)(P + ((size_t)((tile * 4 + s) * 32 + lane)) * 16);
}

static __device__ __forceinline__ v8f wmma_bf16(v16bf a, v16bf b, v8f c) {
  return __builtin_amdgcn_wmma_f32_16x16x32_bf16(
      /*neg_a=*/false, a, /*neg_b=*/false, b,
      /*c_mod=*/(short)0, c, /*reuse_a=*/false, /*reuse_b=*/false);
}

// D (16x16 f32) -> LDS: lane -> N = lane&15, M = v + 8*(lane>>4)
static __device__ __forceinline__ void store_tile(float* S, int stride, int nbase, v8f acc) {
  const int lane = threadIdx.x & 31;
  const int nn = nbase + (lane & 15);
  const int mh = (lane >> 4) * 8;
  #pragma unroll
  for (int v = 0; v < 8; ++v) S[(mh + v) * stride + nn] = acc[v];
}

// ---------------------------------------------------------------------------
// One-shot weight pre-pack: f32 row-major -> bf16 B-fragment layout.
// B[k][n] = W[gbase+n][k]; lane -> N = lane&15, half = lane>>4;
//   elems 0..7  = W[g][s*32 + half*8 + j]
//   elems 8..15 = W[g][s*32 + 16 + half*8 + j]
// ---------------------------------------------------------------------------
__global__ __launch_bounds__(256)
void prepack_kernel(const float* __restrict__ W_iou, const float* __restrict__ U_iou,
                    const float* __restrict__ W_f,   const float* __restrict__ U_f,
                    __bf16* __restrict__ out) {
  const int idx = blockIdx.x * 256 + threadIdx.x;   // 0 .. 64*4*32-1 = 8191
  const int lane = idx & 31;
  const int s    = (idx >> 5) & 3;
  const int tile = idx >> 7;
  const float* W;
  int gbase;
  if (tile < 24)      { W = W_iou; gbase = tile * 16; }
  else if (tile < 48) { W = U_iou; gbase = (tile - 24) * 16; }
  else if (tile < 56) { W = W_f;   gbase = (tile - 48) * 16; }
  else                { W = U_f;   gbase = (tile - 56) * 16; }
  const int g = gbase + (lane & 15);
  const float* p = W + (size_t)g * HID + s * 32 + (lane >> 4) * 8;
  __bf16* o = out + (size_t)idx * 16;
  #pragma unroll
  for (int j = 0; j < 8; ++j) { o[j] = (__bf16)p[j]; o[j + 8] = (__bf16)p[j + 16]; }
}

// ---------------------------------------------------------------------------
// Per-level fused kernel: 16 rows (b,node) per workgroup, 8 waves.
// ---------------------------------------------------------------------------
template <bool LEAF>
__global__ __launch_bounds__(256)
void tree_level_kernel(const int*   __restrict__ node_ids,
                       const float* __restrict__ embedding,
                       const float* __restrict__ b_iou,
                       const float* __restrict__ b_f,
                       const __bf16* __restrict__ wp,      // packed weights
                       float* __restrict__ h_buf,
                       float* __restrict__ c_buf,
                       int l, int n,
                       float* __restrict__ oh,
                       float* __restrict__ oc) {
  __shared__ __align__(16) __bf16 sX  [16 * HID];                 // x tile (bf16)
  __shared__ __align__(16) float  sIOU[16 * 384];                 // iou pre-activations
  __shared__ __align__(16) __bf16 sHS [LEAF ? 8 : 16 * HID];      // h_child sum (bf16)
  __shared__ __align__(16) __bf16 sCH0[LEAF ? 8 : 16 * HID];      // child-0 h (bf16)
  __shared__ __align__(16) __bf16 sCH1[LEAF ? 8 : 16 * HID];      // child-1 h (bf16)
  __shared__ __align__(16) float  sFH0[LEAF ? 8 : 16 * HID];      // x*Wf^T + ch0*Uf^T
  __shared__ __align__(16) float  sFH1[LEAF ? 8 : 16 * HID];      // x*Wf^T + ch1*Uf^T

  const int m   = threadIdx.x >> 4;          // row within tile: 0..15
  const int off = (threadIdx.x & 15) * 8;    // 8-element column chunk

  const int r    = blockIdx.x * 16 + m;      // flat row = b * n + nidx
  const int b    = r >> l;
  const int nidx = r & (n - 1);
  const int node = (n - 1) + nidx;           // heap index within tree

  // ---- stage inputs into LDS (f32 -> bf16) -------------------------------
  const int id = node_ids[b * NNODES + node];
  {
    const float* xp = embedding + (size_t)id * HID + off;
    #pragma unroll
    for (int j = 0; j < 8; ++j) sX[m * HID + off + j] = (__bf16)xp[j];
  }
  if constexpr (!LEAF) {
    const float* h0 = h_buf + ((size_t)b * NNODES + 2 * node + 1) * HID + off;
    const float* h1 = h_buf + ((size_t)b * NNODES + 2 * node + 2) * HID + off;
    #pragma unroll
    for (int j = 0; j < 8; ++j) {
      float a0 = h0[j], a1 = h1[j];
      sCH0[m * HID + off + j] = (__bf16)a0;
      sCH1[m * HID + off + j] = (__bf16)a1;
      sHS [m * HID + off + j] = (__bf16)(a0 + a1);
    }
  }
  __syncthreads();

  // ---- WMMA phase --------------------------------------------------------
  // tiles 0..23 : iou columns (384 = 24*16), K=128 (+128 via U_iou if internal)
  // tiles 24..39: per-child forget-gate pre-act (2 children * 8 col-tiles), K=256
  const int wave = threadIdx.x >> 5;
  const int ntiles = LEAF ? 24 : 40;
  for (int t = wave; t < ntiles; t += 8) {
    v8f acc = {};
    if (t < 24) {
      #pragma unroll
      for (int s = 0; s < 4; ++s)
        acc = wmma_bf16(load_a_frag(sX, s * 32), load_b_frag(wp, t, s), acc);
      if constexpr (!LEAF) {
        #pragma unroll
        for (int s = 0; s < 4; ++s)
          acc = wmma_bf16(load_a_frag(sHS, s * 32), load_b_frag(wp, 24 + t, s), acc);
      }
      store_tile(sIOU, 384, t * 16, acc);
    } else if constexpr (!LEAF) {
      const int k  = (t - 24) >> 3;            // child 0/1
      const int ct = (t - 24) & 7;             // column tile within 128
      const __bf16* CH = k ? sCH1 : sCH0;
      float*        FH = k ? sFH1 : sFH0;
      #pragma unroll
      for (int s = 0; s < 4; ++s)
        acc = wmma_bf16(load_a_frag(sX, s * 32), load_b_frag(wp, 48 + ct, s), acc);
      #pragma unroll
      for (int s = 0; s < 4; ++s)
        acc = wmma_bf16(load_a_frag(CH, s * 32), load_b_frag(wp, 56 + ct, s), acc);
      store_tile(FH, HID, ct * 16, acc);
    }
  }
  __syncthreads();

  // ---- gate epilogue (branch-free, TRANS-op activations) -----------------
  float* hrow = h_buf + ((size_t)b * NNODES + node) * HID;
  float* crow = c_buf + ((size_t)b * NNODES + node) * HID;
  const float* cc0 = nullptr;
  const float* cc1 = nullptr;
  if constexpr (!LEAF) {
    cc0 = c_buf + ((size_t)b * NNODES + 2 * node + 1) * HID;
    cc1 = c_buf + ((size_t)b * NNODES + 2 * node + 2) * HID;
  }
  #pragma unroll
  for (int j = 0; j < 8; ++j) {
    const int k = off + j;
    float iv = sIOU[m * 384 + k]       + b_iou[k];
    float ov = sIOU[m * 384 + 128 + k] + b_iou[128 + k];
    float uv = sIOU[m * 384 + 256 + k] + b_iou[256 + k];
    float csum = 0.0f;
    if constexpr (!LEAF) {
      float bf = b_f[k];
      float f0 = sigm(sFH0[m * HID + k] + bf);
      float f1 = sigm(sFH1[m * HID + k] + bf);
      csum = f0 * cc0[k] + f1 * cc1[k];
    }
    float cv = sigm(iv) * tanh_fast(uv) + csum;
    float hv = sigm(ov) * tanh_fast(cv);
    hrow[k] = hv;
    crow[k] = cv;
    if (oh) { oh[b * HID + k] = hv; oc[b * HID + k] = cv; }  // root level only
  }
}

extern "C" void kernel_launch(void* const* d_in, const int* in_sizes, int n_in,
                              void* d_out, int out_size, void* d_ws, size_t ws_size,
                              hipStream_t stream) {
  (void)in_sizes; (void)n_in; (void)out_size; (void)ws_size;
  const int*   node_ids  = (const int*)  d_in[0];
  const float* embedding = (const float*)d_in[1];
  const float* W_iou     = (const float*)d_in[2];
  const float* b_iou     = (const float*)d_in[3];
  const float* U_iou     = (const float*)d_in[4];
  const float* W_f       = (const float*)d_in[5];
  const float* b_f       = (const float*)d_in[6];
  const float* U_f       = (const float*)d_in[7];
  float* out = (float*)d_out;

  float*  h_buf = (float*)d_ws;
  float*  c_buf = h_buf + (size_t)NB * NNODES * HID;
  __bf16* wp    = (__bf16*)(c_buf + (size_t)NB * NNODES * HID);

  // one-shot weight pre-pack (bf16, B-fragment layout; 256 KB, L2-resident)
  prepack_kernel<<<(NPACK_TILES * 4 * 32) / 256, 256, 0, stream>>>(
      W_iou, U_iou, W_f, U_f, wp);

  for (int l = 7; l >= 0; --l) {
    const int n      = 1 << l;
    const int blocks = (NB * n) / 16;          // 16 rows per workgroup
    float* oh = (l == 0) ? out : nullptr;
    float* oc = (l == 0) ? out + (size_t)NB * HID : nullptr;
    if (l == 7) {
      tree_level_kernel<true><<<blocks, 256, 0, stream>>>(
          node_ids, embedding, b_iou, b_f, wp, h_buf, c_buf, l, n, oh, oc);
    } else {
      tree_level_kernel<false><<<blocks, 256, 0, stream>>>(
          node_ids, embedding, b_iou, b_f, wp, h_buf, c_buf, l, n, oh, oc);
    }
  }
}
